// DeformableTransformer_70050916597909
// MI455X (gfx1250) — matrix operand order repaired and checked by reading, not compiled, not run on hardware
//
#include <hip/hip_runtime.h>

// ---------------------------------------------------------------------------
// Deformable transformer encoder, MI455X / gfx1250 (wave32, WMMA).
// GEMMs: v_wmma_f32_16x16x32_f16, B-panel staged in LDS via
// global_load_async_to_lds_b128 (ASYNCcnt) with ds fallback.
// ---------------------------------------------------------------------------

#define Bc   2
#define Sc   13294
#define Dc   256
#define Hc   8
#define DHc  32
#define LVLc 4
#define Pc   4
#define DFFc 1024
#define NLc  6
#define Mc   (Bc * Sc)          // 26588 rows

typedef __attribute__((ext_vector_type(16))) _Float16 v16h;
typedef __attribute__((ext_vector_type(8)))  _Float16 v8h;
typedef __attribute__((ext_vector_type(8)))  float    v8f;
typedef __attribute__((ext_vector_type(4)))  int      v4i;

#if defined(__has_builtin)
#if __has_builtin(__builtin_amdgcn_global_load_async_to_lds_b128) && \
    __has_builtin(__builtin_amdgcn_s_wait_asynccnt)
#define USE_ASYNC_LDS 1
#endif
#endif

#define AS_GLOBAL __attribute__((address_space(1)))
#define AS_LOCAL  __attribute__((address_space(3)))

__device__ __forceinline__ v16h combine16(v8h lo, v8h hi) {
    v16h r;
#pragma unroll
    for (int e = 0; e < 8; ++e) { r[e] = lo[e]; r[e + 8] = hi[e]; }
    return r;
}

#if defined(USE_ASYNC_LDS)
__device__ __forceinline__ void async_copy16B(const _Float16* g, _Float16* l) {
    __builtin_amdgcn_global_load_async_to_lds_b128(
        (AS_GLOBAL v4i*)(void*)g,
        (AS_LOCAL  v4i*)(void*)l, 0, 0);
}
#endif

// ---------------------------------------------------------------------------
// WMMA GEMM:  C[M,N] = A[M,K] @ W[K,N] + bias,  A f16 row-major,
// Wt = W^T f16 (N rows of K).  Block = 256 threads = 8 waves, tile 64M x 32N.
// B panel (32 x 64-K chunk) double-buffered in LDS, shared by the 4 M-waves.
// Row stride 72 halves (144 B): rows stay 16B-aligned and ds_load_b128 reads
// across lanes are bank-conflict free (36 DW stride, 9 coprime with 16).
// ---------------------------------------------------------------------------
template <bool RELU>
__global__ __launch_bounds__(256)
void wmma_gemm_kernel(const _Float16* __restrict__ A,
                      const _Float16* __restrict__ Wt,
                      const float* __restrict__ bias,
                      float* __restrict__ C,
                      _Float16* __restrict__ Ch,
                      int M, int N, int K) {
    constexpr int KC   = 64;
    constexpr int BPAD = 72;
    __shared__ _Float16 Bs[2][32][BPAD];

    const int lane      = threadIdx.x & 31;
    const int wave      = threadIdx.x >> 5;
    const int wm        = wave >> 1;          // 0..3
    const int wn        = wave & 1;           // 0..1
    const int tileM     = blockIdx.y * 64 + wm * 16;
    const int tileN     = blockIdx.x * 32 + wn * 16;
    const int half_lane = lane >> 4;          // 0 or 1
    const int l15       = lane & 15;

    // A-fragment: lanes 0-15 row M=l15, K = k..k+7 / k+16..k+23;
    //             lanes 16-31 same rows,  K = k+8..k+15 / k+24..k+31.
    int arow = tileM + l15;
    if (arow >= M) arow = M - 1;              // clamp; stores guarded below
    const _Float16* Arow  = A + (size_t)arow * K;
    const int       a_off = half_lane * 8;

    // Cooperative B staging: 256 threads move 32 rows x 64 halves (16B each).
    const int srow = threadIdx.x >> 3;        // 0..31
    const int sseg = (threadIdx.x & 7) * 8;   // 0,8,...,56
    const _Float16* Sg = Wt + (size_t)(blockIdx.x * 32 + srow) * K + sseg;

    v8f acc = {};
    const int nc = K / KC;

    // prologue stage of chunk 0
    {
#if defined(USE_ASYNC_LDS)
        async_copy16B(Sg, &Bs[0][srow][sseg]);
#else
        *(v8h*)&Bs[0][srow][sseg] = *(const v8h*)Sg;
#endif
    }

    for (int c = 0; c < nc; ++c) {
        const int k0 = c * KC;
#if defined(USE_ASYNC_LDS)
        __builtin_amdgcn_s_wait_asynccnt(0);
#endif
        __syncthreads();                      // Bs[c&1] ready; prev reads done

        if (c + 1 < nc) {
            const _Float16* g = Sg + k0 + KC;
#if defined(USE_ASYNC_LDS)
            async_copy16B(g, &Bs[(c + 1) & 1][srow][sseg]);
#else
            *(v8h*)&Bs[(c + 1) & 1][srow][sseg] = *(const v8h*)g;
#endif
            // keep the A stream ahead of the WMMAs
            __builtin_prefetch(Arow + k0 + 2 * KC, 0, 0);
        }

        const _Float16* brow = &Bs[c & 1][wn * 16 + l15][0];
#pragma unroll
        for (int kk = 0; kk < KC; kk += 32) {
            v8h a0 = *(const v8h*)(Arow + k0 + kk + a_off);
            v8h a1 = *(const v8h*)(Arow + k0 + kk + a_off + 16);
            v8h b0 = *(const v8h*)(brow + kk + half_lane * 8);
            v8h b1 = *(const v8h*)(brow + kk + half_lane * 8 + 16);
            v16h a = combine16(a0, a1);
            v16h b = combine16(b0, b1);
            acc = __builtin_amdgcn_wmma_f32_16x16x32_f16(
                false, a, false, b, (short)0, acc, false, false);
        }
    }

    const float bv = bias ? bias[tileN + l15] : 0.0f;
#pragma unroll
    for (int r = 0; r < 8; ++r) {
        const int row = tileM + r + 8 * half_lane;   // C layout: VGPR r -> M=r / r+8
        if (row < M) {
            float v = acc[r] + bv;
            if (RELU) v = fmaxf(v, 0.0f);
            const size_t idx = (size_t)row * N + tileN + l15;
            if (C)  C[idx]  = v;
            if (Ch) Ch[idx] = (_Float16)v;
        }
    }
}

// ---------------------------------------------------------------------------
// Small elementwise / prep kernels
// ---------------------------------------------------------------------------
__global__ void copy_cvt_kernel(const float* __restrict__ src,
                                float* __restrict__ dst,
                                _Float16* __restrict__ dsth, int n) {
    int i = blockIdx.x * blockDim.x + threadIdx.x;
    if (i >= n) return;
    float v = src[i];
    dst[i]  = v;
    dsth[i] = (_Float16)v;
}

__global__ void add_pos_kernel(const float* __restrict__ cur,
                               const float* __restrict__ pos,
                               _Float16* __restrict__ qh, int n) {
    int i = blockIdx.x * blockDim.x + threadIdx.x;
    if (i >= n) return;
    qh[i] = (_Float16)(cur[i] + pos[i]);
}

// reference points: ref[bs][lvl][c] = base(b, s, c) * valid_ratios[b][lvl][c]
__global__ void ref_points_kernel(const float* __restrict__ vr,
                                  float* __restrict__ ref) {
    int i = blockIdx.x * blockDim.x + threadIdx.x;     // over B*S
    if (i >= Mc) return;
    const int b = i / Sc;
    const int s = i % Sc;
    const int HH[4] = {100, 50, 25, 13};
    const int ST[4] = {0, 10000, 12500, 13125};
    int lq = 3;
#pragma unroll
    for (int l = 0; l < 4; ++l)
        if (s >= ST[l] && s < ST[l] + HH[l] * HH[l]) lq = l;
    const int rel = s - ST[lq];
    const int Wl  = HH[lq];
    const float ry = ((float)(rel / Wl) + 0.5f) / (vr[b * 8 + lq * 2 + 1] * (float)HH[lq]);
    const float rx = ((float)(rel % Wl) + 0.5f) / (vr[b * 8 + lq * 2 + 0] * (float)Wl);
#pragma unroll
    for (int l = 0; l < 4; ++l) {
        ref[(size_t)i * 8 + l * 2 + 0] = rx * vr[b * 8 + l * 2 + 0];
        ref[(size_t)i * 8 + l * 2 + 1] = ry * vr[b * 8 + l * 2 + 1];
    }
}

// W (K x N, f32) -> Wt (N x K, f16)
__global__ void transpose_cvt_kernel(const float* __restrict__ W,
                                     _Float16* __restrict__ Wt,
                                     int K, int N) {
    int i = blockIdx.x * blockDim.x + threadIdx.x;
    if (i >= K * N) return;
    const int n = i / K;
    const int k = i % K;
    Wt[i] = (_Float16)W[(size_t)k * N + n];
}

// softmax over groups of 16 (LVL*P) — one thread per (b, s, h)
__global__ void softmax16_kernel(float* __restrict__ a, int total) {
    int i = blockIdx.x * blockDim.x + threadIdx.x;
    if (i >= total) return;
    float* p = a + (size_t)i * 16;
    float mx = -3.4e38f;
#pragma unroll
    for (int j = 0; j < 16; ++j) mx = fmaxf(mx, p[j]);
    float s = 0.0f;
#pragma unroll
    for (int j = 0; j < 16; ++j) { float e = __expf(p[j] - mx); p[j] = e; s += e; }
    const float inv = 1.0f / s;
#pragma unroll
    for (int j = 0; j < 16; ++j) p[j] *= inv;
}

// ---------------------------------------------------------------------------
// Deformable attention sampler: one wave per (b*s, head); lane = channel (DH=32).
// Sample coords / weights are wave-uniform -> scalarized; value gathers are
// fully coalesced (32 consecutive floats).
// ---------------------------------------------------------------------------
__global__ __launch_bounds__(256)
void ms_deform_kernel(const float* __restrict__ value,
                      const float* __restrict__ ref,
                      const float* __restrict__ off,
                      const float* __restrict__ attn,
                      _Float16* __restrict__ outh) {
    const int wid = blockIdx.x * (blockDim.x >> 5) + (threadIdx.x >> 5);
    if (wid >= Mc * Hc) return;
    const int bs   = wid >> 3;
    const int h    = wid & 7;
    const int b    = bs / Sc;
    const int lane = threadIdx.x & 31;

    const float* refp = ref  + (size_t)bs * 8;
    const float* offp = off  + (size_t)bs * 256 + h * 32;   // (lvl*4+p)*2+c
    const float* ap   = attn + (size_t)bs * 128 + h * 16;   // lvl*4+p
    const int HH[4] = {100, 50, 25, 13};

    float acc = 0.0f;
    int start = 0;
#pragma unroll
    for (int lvl = 0; lvl < 4; ++lvl) {
        const int Hl = HH[lvl];
        const int Wl = HH[lvl];
        const float fW = (float)Wl, fH = (float)Hl;
        const float* vb = value + ((size_t)(b * Sc + start)) * Dc + h * DHc + lane;
#pragma unroll
        for (int p = 0; p < 4; ++p) {
            const float lx = refp[lvl * 2 + 0] + offp[(lvl * 4 + p) * 2 + 0] / fW;
            const float ly = refp[lvl * 2 + 1] + offp[(lvl * 4 + p) * 2 + 1] / fH;
            const float x  = lx * fW - 0.5f;
            const float y  = ly * fH - 0.5f;
            const float x0 = floorf(x), y0 = floorf(y);
            const float wx1 = x - x0, wy1 = y - y0;
            const float wx0 = 1.0f - wx1, wy0 = 1.0f - wy1;
            const int xi = (int)x0, yi = (int)y0;
            const float a = ap[lvl * 4 + p];

            float sv = 0.0f;
#pragma unroll
            for (int cy = 0; cy < 2; ++cy) {
#pragma unroll
                for (int cx = 0; cx < 2; ++cx) {
                    const int xx = xi + cx, yy = yi + cy;
                    if (xx >= 0 && xx < Wl && yy >= 0 && yy < Hl) {
                        const float w = (cx ? wx1 : wx0) * (cy ? wy1 : wy0);
                        sv += w * vb[(size_t)(yy * Wl + xx) * Dc];
                    }
                }
            }
            acc += a * sv;
        }
        start += Hl * Wl;
    }
    outh[(size_t)bs * Dc + h * DHc + lane] = (_Float16)acc;
}

// ---------------------------------------------------------------------------
// Fused residual + LayerNorm: one wave per row of 256; lane owns 8 elements.
// Writes f32 (next residual) and f16 (next GEMM operand).
// ---------------------------------------------------------------------------
__global__ __launch_bounds__(256)
void add_ln_kernel(const float* __restrict__ x,
                   const float* __restrict__ res,
                   const float* __restrict__ g,
                   const float* __restrict__ bta,
                   float* __restrict__ out,
                   _Float16* __restrict__ outh, int M) {
    const int row = blockIdx.x * (blockDim.x >> 5) + (threadIdx.x >> 5);
    if (row >= M) return;
    const int lane = threadIdx.x & 31;
    const float* xr = x   + (size_t)row * Dc + lane * 8;
    const float* rr = res + (size_t)row * Dc + lane * 8;

    float v[8];
    float s = 0.0f;
#pragma unroll
    for (int i = 0; i < 8; ++i) { v[i] = xr[i] + rr[i]; s += v[i]; }
#pragma unroll
    for (int m = 16; m > 0; m >>= 1) s += __shfl_xor(s, m, 32);
    const float mean = s * (1.0f / Dc);

    float var = 0.0f;
#pragma unroll
    for (int i = 0; i < 8; ++i) { float d = v[i] - mean; var += d * d; }
#pragma unroll
    for (int m = 16; m > 0; m >>= 1) var += __shfl_xor(var, m, 32);
    const float inv = rsqrtf(var * (1.0f / Dc) + 1e-5f);

    float*     orow  = out  + (size_t)row * Dc + lane * 8;
    _Float16*  ohrow = outh + (size_t)row * Dc + lane * 8;
    const float* gp  = g   + lane * 8;
    const float* bp  = bta + lane * 8;
#pragma unroll
    for (int i = 0; i < 8; ++i) {
        float o   = (v[i] - mean) * inv * gp[i] + bp[i];
        orow[i]   = o;
        ohrow[i]  = (_Float16)o;
    }
}

// ---------------------------------------------------------------------------
// Host orchestration
// ---------------------------------------------------------------------------
extern "C" void kernel_launch(void* const* d_in, const int* in_sizes, int n_in,
                              void* d_out, int out_size, void* d_ws, size_t ws_size,
                              hipStream_t stream) {
    (void)in_sizes; (void)n_in; (void)out_size; (void)ws_size;

    const float* src    = (const float*)d_in[0];
    const float* pos    = (const float*)d_in[1];
    const float* vr     = (const float*)d_in[2];
    const float* so_w   = (const float*)d_in[3];
    const float* so_b   = (const float*)d_in[4];
    const float* aw_w   = (const float*)d_in[5];
    const float* aw_b   = (const float*)d_in[6];
    const float* vp_w   = (const float*)d_in[7];
    const float* vp_b   = (const float*)d_in[8];
    const float* op_w   = (const float*)d_in[9];
    const float* op_b   = (const float*)d_in[10];
    const float* ln1_g  = (const float*)d_in[11];
    const float* ln1_b  = (const float*)d_in[12];
    const float* f1_w   = (const float*)d_in[13];
    const float* f1_b   = (const float*)d_in[14];
    const float* f2_w   = (const float*)d_in[15];
    const float* f2_b   = (const float*)d_in[16];
    const float* ln2_g  = (const float*)d_in[17];
    const float* ln2_b  = (const float*)d_in[18];

    const int M = Mc;

    // ---- workspace carve-up ----
    char*  base = (char*)d_ws;
    size_t off  = 0;
    auto carve = [&](size_t bytes) -> char* {
        char* p = base + off;
        off = (off + bytes + 255) & ~(size_t)255;
        return p;
    };
    float*    refb   = (float*)   carve((size_t)M * 8 * 4);
    float*    cur    = (float*)   carve((size_t)M * Dc * 4);
    _Float16* cur_h  = (_Float16*)carve((size_t)M * Dc * 2);
    _Float16* q_h    = (_Float16*)carve((size_t)M * Dc * 2);
    float*    offb   = (float*)   carve((size_t)M * 256 * 4);
    float*    attnb  = (float*)   carve((size_t)M * 128 * 4);
    float*    valb   = (float*)   carve((size_t)M * Dc * 4);
    _Float16* samp_h = (_Float16*)carve((size_t)M * Dc * 2);
    float*    projb  = (float*)   carve((size_t)M * Dc * 4);
    _Float16* fmid_h = (_Float16*)carve((size_t)M * DFFc * 2);
    float*    ffout  = (float*)   carve((size_t)M * Dc * 4);
    _Float16* wt_so  = (_Float16*)carve((size_t)NLc * Dc * 256 * 2);
    _Float16* wt_aw  = (_Float16*)carve((size_t)NLc * Dc * 128 * 2);
    _Float16* wt_vp  = (_Float16*)carve((size_t)NLc * Dc * Dc * 2);
    _Float16* wt_op  = (_Float16*)carve((size_t)NLc * Dc * Dc * 2);
    _Float16* wt_f1  = (_Float16*)carve((size_t)NLc * Dc * DFFc * 2);
    _Float16* wt_f2  = (_Float16*)carve((size_t)NLc * DFFc * Dc * 2);

    auto tcvt = [&](const float* W, _Float16* Wt, int K, int N) {
        int tot = K * N;
        transpose_cvt_kernel<<<(tot + 255) / 256, 256, 0, stream>>>(W, Wt, K, N);
    };
    for (int l = 0; l < NLc; ++l) {
        tcvt(so_w + (size_t)l * Dc * 256,  wt_so + (size_t)l * Dc * 256,  Dc,   256);
        tcvt(aw_w + (size_t)l * Dc * 128,  wt_aw + (size_t)l * Dc * 128,  Dc,   128);
        tcvt(vp_w + (size_t)l * Dc * Dc,   wt_vp + (size_t)l * Dc * Dc,   Dc,   Dc);
        tcvt(op_w + (size_t)l * Dc * Dc,   wt_op + (size_t)l * Dc * Dc,   Dc,   Dc);
        tcvt(f1_w + (size_t)l * Dc * DFFc, wt_f1 + (size_t)l * Dc * DFFc, Dc,   DFFc);
        tcvt(f2_w + (size_t)l * DFFc * Dc, wt_f2 + (size_t)l * DFFc * Dc, DFFc, Dc);
    }

    const int nElem = M * Dc;
    copy_cvt_kernel<<<(nElem + 255) / 256, 256, 0, stream>>>(src, cur, cur_h, nElem);
    ref_points_kernel<<<(M + 255) / 256, 256, 0, stream>>>(vr, refb);

    auto gemm = [&](const _Float16* A, const _Float16* Wt, const float* bias,
                    float* C, _Float16* Ch, int N, int K, bool relu) {
        dim3 grid(N / 32, (M + 63) / 64);
        if (relu)
            wmma_gemm_kernel<true ><<<grid, 256, 0, stream>>>(A, Wt, bias, C, Ch, M, N, K);
        else
            wmma_gemm_kernel<false><<<grid, 256, 0, stream>>>(A, Wt, bias, C, Ch, M, N, K);
    };

    for (int l = 0; l < NLc; ++l) {
        // q = cur + pos (f16 for the Q-side GEMMs)
        add_pos_kernel<<<(nElem + 255) / 256, 256, 0, stream>>>(cur, pos, q_h, nElem);

        // sampling offsets, attention logits, value projection
        gemm(q_h,   wt_so + (size_t)l * Dc * 256, so_b + (size_t)l * 256, offb,  nullptr, 256, Dc, false);
        gemm(q_h,   wt_aw + (size_t)l * Dc * 128, aw_b + (size_t)l * 128, attnb, nullptr, 128, Dc, false);
        softmax16_kernel<<<(M * Hc + 255) / 256, 256, 0, stream>>>(attnb, M * Hc);
        gemm(cur_h, wt_vp + (size_t)l * Dc * Dc,  vp_b + (size_t)l * Dc,  valb,  nullptr, Dc,  Dc, false);

        // deformable bilinear sampling -> f16 directly for the output projection
        ms_deform_kernel<<<(M * Hc + 7) / 8, 256, 0, stream>>>(valb, refb, offb, attnb, samp_h);

        // output projection + residual + LN1
        gemm(samp_h, wt_op + (size_t)l * Dc * Dc, op_b + (size_t)l * Dc, projb, nullptr, Dc, Dc, false);
        add_ln_kernel<<<(M + 7) / 8, 256, 0, stream>>>(cur, projb,
                                                       ln1_g + (size_t)l * Dc, ln1_b + (size_t)l * Dc,
                                                       cur, cur_h, M);

        // FFN: relu(x @ f1 + b1) @ f2 + b2, residual + LN2
        gemm(cur_h,  wt_f1 + (size_t)l * Dc * DFFc, f1_b + (size_t)l * DFFc, nullptr, fmid_h, DFFc, Dc,  true);
        gemm(fmid_h, wt_f2 + (size_t)l * DFFc * Dc, f2_b + (size_t)l * Dc,   ffout,   nullptr, Dc,   DFFc, false);
        add_ln_kernel<<<(M + 7) / 8, 256, 0, stream>>>(cur, ffout,
                                                       ln2_g + (size_t)l * Dc, ln2_b + (size_t)l * Dc,
                                                       cur, cur_h, M);
    }

    (void)hipMemcpyAsync(d_out, cur, (size_t)M * Dc * 4, hipMemcpyDeviceToDevice, stream);
}
